// TabNetModel_69337952026712
// MI455X (gfx1250) — matrix operand order repaired and checked by reading, not compiled, not run on hardware
//
#include <hip/hip_runtime.h>

// ---------------------------------------------------------------------------
// TabNet inference, gfx1250 (MI455X), wave32 + WMMA bf16 + async LDS staging.
//
// Live dataflow only (attention branch of the reference is provably dead):
//   x' = BN0(x)
//   sh = GLU(BN(x' @ shW))                         [B,128]
//   agg = sum_s GLU(BN(sh @ stW[s]))[:, :64]       [B,64]
//   out = agg @ fW + fb                            [B,2]
// BNs folded into bf16 weight scales + f32 biases. Weights staged into LDS
// once per block (double-buffered, async-to-LDS) and shared by all 8 waves,
// cutting weight L2 traffic ~8x so the kernel stays on the 23.3 TB/s HBM
// roofline (~268 MB of x => ~11.5 us floor).
// ---------------------------------------------------------------------------

#define TB_EPS 1e-5f
#define USE_ASYNC 1   // gfx1250 GLOBAL_LOAD_ASYNC_TO_LDS_B128 staging path

typedef __attribute__((ext_vector_type(16))) __bf16        v16bf;
typedef __attribute__((ext_vector_type(8)))  float         v8f;
typedef __attribute__((ext_vector_type(4)))  unsigned int  u32x4;

union Frag16 { v16bf v; u32x4 q[2]; };

__device__ __forceinline__ unsigned short f2bf(float f) {
  return __builtin_bit_cast(unsigned short, (__bf16)f);
}

// B fragment (32x16, 16-bit): lane holds 16 contiguous K at k = g*16.
__device__ __forceinline__ v16bf load_fragB(const unsigned short* p) {
  Frag16 f;
  f.q[0] = *(const u32x4*)(p);
  f.q[1] = *(const u32x4*)(p + 8);
  return f.v;
}
// A fragment (16x32, 16-bit): lane holds K chunks {g*16..g*16+7} and {g*16+8..}.
__device__ __forceinline__ v16bf load_fragA(const unsigned short* p0,
                                            const unsigned short* p1) {
  Frag16 f;
  f.q[0] = *(const u32x4*)(p0);
  f.q[1] = *(const u32x4*)(p1);
  return f.v;
}

__device__ __forceinline__ v8f wmma_bf16(v16bf a, v16bf b, v8f c) {
  return __builtin_amdgcn_wmma_f32_16x16x32_bf16(
      false, a, false, b, (short)0, c, false, false);
}

// v_exp_f32 + v_rcp_f32 (avoid the IEEE v_div_scale expansion seen in rnd 1)
__device__ __forceinline__ float sigmoidf_fast(float x) {
  return __builtin_amdgcn_rcpf(1.0f + __expf(-x));
}

// ---------------------------------------------------------------------------
// Prep 1: W1t[n][k] = shW[k][n] * s_sh[n]   (bf16, 256 x 512, K-contiguous)
//         b1[n]     = sh_b[n] - sh_m[n]*s_sh[n];  s0/t0 = BN0 scale/shift
// ---------------------------------------------------------------------------
__global__ __launch_bounds__(256) void tabnet_prep_w1(
    const float* __restrict__ shW, const float* __restrict__ sh_g,
    const float* __restrict__ sh_b, const float* __restrict__ sh_m,
    const float* __restrict__ sh_v,
    const float* __restrict__ bn0_g, const float* __restrict__ bn0_b,
    const float* __restrict__ bn0_m, const float* __restrict__ bn0_v,
    unsigned short* __restrict__ W1t, float* __restrict__ b1,
    float* __restrict__ s0, float* __restrict__ t0) {
  int idx = blockIdx.x * 256 + threadIdx.x;       // 512 blocks -> 131072
  int n = idx >> 9, k = idx & 511;
  float s = sh_g[n] * rsqrtf(sh_v[n] + TB_EPS);
  W1t[(size_t)n * 512 + k] = f2bf(shW[(size_t)k * 256 + n] * s);
  if (k == 0) b1[n] = sh_b[n] - sh_m[n] * s;
  if (n == 0) {
    float sk = bn0_g[k] * rsqrtf(bn0_v[k] + TB_EPS);
    s0[k] = sk;
    t0[k] = bn0_b[k] - bn0_m[k] * sk;
  }
}

// ---------------------------------------------------------------------------
// Prep 2: pack live stW columns {0..63, 128..191}, BN-scaled, transposed:
//   W2t[s][n][k] = stW[s][k][ncol] * s_st[s][ncol],  ncol = n<64 ? n : n+64
// ---------------------------------------------------------------------------
__global__ __launch_bounds__(256) void tabnet_prep_w2(
    const float* __restrict__ stW, const float* __restrict__ st_g,
    const float* __restrict__ st_b, const float* __restrict__ st_m,
    const float* __restrict__ st_v,
    unsigned short* __restrict__ W2t, float* __restrict__ b2) {
  int idx = blockIdx.x * 256 + threadIdx.x;       // 192 blocks -> 49152
  int s = idx >> 14;
  int n = (idx >> 7) & 127;
  int k = idx & 127;
  int ncol = (n < 64) ? n : (n + 64);
  float sc = st_g[s * 256 + ncol] * rsqrtf(st_v[s * 256 + ncol] + TB_EPS);
  W2t[((size_t)s * 128 + n) * 128 + k] =
      f2bf(stW[((size_t)s * 128 + k) * 256 + ncol] * sc);
  if (k == 0) b2[s * 128 + n] = st_b[s * 256 + ncol] - st_m[s * 256 + ncol] * sc;
}

// ---------------------------------------------------------------------------
// Fused main kernel: 256 threads = 8 wave32; each wave owns 16 rows.
// LDS map (bytes):
//   [0      , 133120) xs   : 8 waves x 16 x 520 bf16 (x' tiles; row pad ->
//                            stride 260 dw == 4 mod 64 banks); aliased as
//                            per-wave agg staging at the end
//   [133120 , 167936) shl  : 8 waves x 16 x 136 bf16 (sh tiles)
//   [167936 , 237568) wtile: 2 x 34816 B weight-tile double buffer
// ---------------------------------------------------------------------------
#define XS_STRIDE 520
#define SH_STRIDE 136
#define AG_STRIDE 68
#define SHL_OFF   133120
#define WT_OFF    167936
#define WT_BYTES  34816
#define SMEM_BYTES 237568

__global__ __launch_bounds__(256) void tabnet_main(
    const float* __restrict__ x,
    const float* __restrict__ s0, const float* __restrict__ t0,
    const unsigned short* __restrict__ W1t, const float* __restrict__ b1,
    const unsigned short* __restrict__ W2t, const float* __restrict__ b2,
    const float* __restrict__ fW, const float* __restrict__ fb,
    float* __restrict__ out) {
  extern __shared__ char smem[];
  const int tid  = threadIdx.x;
  const int lane = tid & 31;
  const int wave = tid >> 5;
  const int g    = lane >> 4;           // half-wave group (K-chunk select)
  const int ln   = lane & 15;           // M/N index within fragment
  const int rowTile = blockIdx.x * 128 + wave * 16;

  unsigned short* xs  = (unsigned short*)smem + (size_t)wave * 16 * XS_STRIDE;
  unsigned short* shl = (unsigned short*)(smem + SHL_OFF) +
                        (size_t)wave * 16 * SH_STRIDE;
  float* aggl = (float*)xs;             // per-wave alias; xs dead by stage 3

  // 16B-unit copy helper: global -> LDS (async-to-LDS on CDNA5)
  auto copy16 = [&](unsigned short* dst, const unsigned short* src) {
#if USE_ASYNC
    asm volatile("global_load_async_to_lds_b128 %0, %1, off"
                 :: "v"((unsigned)(size_t)dst),
                    "v"((unsigned long long)(size_t)src)
                 : "memory");
#else
    *(u32x4*)dst = *(const u32x4*)src;
#endif
  };
  // Stage GEMM1 weight tile np (rows np*16..+15 and 128+np*16..+15) -> buf b
  auto stage1 = [&](int np, int b) {
    unsigned short* wt = (unsigned short*)(smem + WT_OFF + b * WT_BYTES);
#pragma unroll
    for (int i = 0; i < 8; ++i) {
      int u = tid + i * 256;            // 2048 x 16B units (32 rows x 1KB)
      int r = u >> 6, cu = u & 63;
      int grow = np * 16 + ((r < 16) ? r : (112 + r));
      copy16(wt + r * XS_STRIDE + cu * 8, W1t + (size_t)grow * 512 + cu * 8);
    }
  };
  // Stage GEMM2 weight tile for step s (128 rows x 128 K) -> buf b
  auto stage2 = [&](int s, int b) {
    unsigned short* wt = (unsigned short*)(smem + WT_OFF + b * WT_BYTES);
#pragma unroll
    for (int i = 0; i < 8; ++i) {
      int u = tid + i * 256;            // 2048 x 16B units (128 rows x 256B)
      int r = u >> 4, cu = u & 15;
      copy16(wt + r * SH_STRIDE + cu * 8,
             W2t + ((size_t)s * 128 + r) * 128 + cu * 8);
    }
  };

  // ---- Stage 0: load x tile, apply BN0, convert to bf16 into LDS ----------
  stage1(0, 0);                         // overlap weight prefetch with x load
  for (int r = 0; r < 16; ++r) {
    const float* xr = x + (size_t)(rowTile + r) * 512;
    if (rowTile + 128 + r < 131072)
      __builtin_prefetch(xr + (size_t)128 * 512, 0, 0);   // global_prefetch_b8
#pragma unroll
    for (int it = 0; it < 4; ++it) {
      int c = (it * 32 + lane) * 4;
      float4 v = *(const float4*)(xr + c);
      float4 s = *(const float4*)(s0 + c);
      float4 t = *(const float4*)(t0 + c);
      uint2 pk;
      pk.x = (unsigned int)f2bf(v.x * s.x + t.x) |
             ((unsigned int)f2bf(v.y * s.y + t.y) << 16);
      pk.y = (unsigned int)f2bf(v.z * s.z + t.z) |
             ((unsigned int)f2bf(v.w * s.w + t.w) << 16);
      *(uint2*)(xs + r * XS_STRIDE + c) = pk;
    }
  }
  asm volatile("s_wait_asynccnt 0x0" ::: "memory");
  __syncthreads();

  const unsigned short* arow = xs + ln * XS_STRIDE;

  // ---- Stage 1: sh = GLU(x' @ W1 + b1), N paired (np, np+8) ---------------
#pragma unroll 1
  for (int np = 0; np < 8; ++np) {
    if (np + 1 < 8) stage1(np + 1, (np + 1) & 1);
    const unsigned short* wt =
        (unsigned short*)(smem + WT_OFF + (np & 1) * WT_BYTES);
    float bb1 = b1[np * 16 + ln];
    float bb2 = b1[np * 16 + 128 + ln];
    v8f acc1, acc2;
#pragma unroll
    for (int r = 0; r < 8; ++r) { acc1[r] = bb1; acc2[r] = bb2; }
    const unsigned short* brow1 = wt + ln * XS_STRIDE;          // d1 rows 0..15
    const unsigned short* brow2 = wt + (16 + ln) * XS_STRIDE;   // d2 rows 16..31
#pragma unroll 4
    for (int ks = 0; ks < 16; ++ks) {
      v16bf a  = load_fragA(arow + ks * 32 + g * 16,
                            arow + ks * 32 + g * 16 + 8);
      v16bf bA = load_fragB(brow1 + ks * 32 + g * 16);
      v16bf bB = load_fragB(brow2 + ks * 32 + g * 16);
      acc1 = wmma_bf16(a, bA, acc1);
      acc2 = wmma_bf16(a, bB, acc2);
    }
#pragma unroll
    for (int r = 0; r < 8; ++r) {
      float gl = acc1[r] * sigmoidf_fast(acc2[r]);
      shl[(r + g * 8) * SH_STRIDE + np * 16 + ln] = f2bf(gl);
    }
    asm volatile("s_wait_asynccnt 0x0" ::: "memory");
    __syncthreads();
  }

  // ---- Stage 2: agg = sum_s GLU(sh @ W2[s] + b2[s])  (live 64 cols) -------
  v8f agg[4];
#pragma unroll
  for (int nt = 0; nt < 4; ++nt)
#pragma unroll
    for (int r = 0; r < 8; ++r) agg[nt][r] = 0.0f;

  const unsigned short* srow = shl + ln * SH_STRIDE;
  stage2(0, 0);
  asm volatile("s_wait_asynccnt 0x0" ::: "memory");
  __syncthreads();
#pragma unroll 1
  for (int s = 0; s < 3; ++s) {
    if (s + 1 < 3) stage2(s + 1, (s + 1) & 1);
    const unsigned short* wt =
        (unsigned short*)(smem + WT_OFF + (s & 1) * WT_BYTES);
    const float* bs = b2 + s * 128;
#pragma unroll
    for (int nt = 0; nt < 4; ++nt) {
      float bb1 = bs[nt * 16 + ln];
      float bb2 = bs[64 + nt * 16 + ln];
      v8f acc1, acc2;
#pragma unroll
      for (int r = 0; r < 8; ++r) { acc1[r] = bb1; acc2[r] = bb2; }
#pragma unroll
      for (int ks = 0; ks < 4; ++ks) {
        v16bf a  = load_fragA(srow + ks * 32 + g * 16,
                              srow + ks * 32 + g * 16 + 8);
        v16bf bd = load_fragB(wt + (nt * 16 + ln) * SH_STRIDE +
                              ks * 32 + g * 16);
        v16bf ba = load_fragB(wt + (64 + nt * 16 + ln) * SH_STRIDE +
                              ks * 32 + g * 16);
        acc1 = wmma_bf16(a, bd, acc1);
        acc2 = wmma_bf16(a, ba, acc2);
      }
#pragma unroll
      for (int r = 0; r < 8; ++r)
        agg[nt][r] += acc1[r] * sigmoidf_fast(acc2[r]);
    }
    asm volatile("s_wait_asynccnt 0x0" ::: "memory");
    __syncthreads();
  }

  // ---- Stage 3: out = agg @ fW + fb  (tiny N=2 projection) ----------------
#pragma unroll
  for (int nt = 0; nt < 4; ++nt)
#pragma unroll
    for (int r = 0; r < 8; ++r)
      aggl[(r + g * 8) * AG_STRIDE + nt * 16 + ln] = agg[nt][r];
  asm volatile("s_wait_dscnt 0x0" ::: "memory");

  {
    int row = ln;          // lanes 0..15 -> o=0, lanes 16..31 -> o=1
    int o   = g;
    float acc = fb[o];
#pragma unroll 8
    for (int c = 0; c < 64; ++c)
      acc += aggl[row * AG_STRIDE + c] * fW[c * 2 + o];
    out[(size_t)(rowTile + row) * 2 + o] = acc;
  }
}

// ---------------------------------------------------------------------------
// Launch
// ---------------------------------------------------------------------------
extern "C" void kernel_launch(void* const* d_in, const int* in_sizes, int n_in,
                              void* d_out, int out_size, void* d_ws,
                              size_t ws_size, hipStream_t stream) {
  const float* x     = (const float*)d_in[0];
  const float* bn0_g = (const float*)d_in[1];
  const float* bn0_b = (const float*)d_in[2];
  const float* bn0_m = (const float*)d_in[3];
  const float* bn0_v = (const float*)d_in[4];
  const float* shW   = (const float*)d_in[5];
  const float* sh_g  = (const float*)d_in[6];
  const float* sh_b  = (const float*)d_in[7];
  const float* sh_m  = (const float*)d_in[8];
  const float* sh_v  = (const float*)d_in[9];
  const float* stW   = (const float*)d_in[10];
  const float* st_g  = (const float*)d_in[11];
  const float* st_b  = (const float*)d_in[12];
  const float* st_m  = (const float*)d_in[13];
  const float* st_v  = (const float*)d_in[14];
  // d_in[15..19]: atW / at_* -> provably dead in the reference output
  const float* fW    = (const float*)d_in[20];
  const float* fb    = (const float*)d_in[21];
  float* out = (float*)d_out;

  char* ws = (char*)d_ws;
  unsigned short* W1t = (unsigned short*)(ws + 0);        // 256*512*2 = 262144
  unsigned short* W2t = (unsigned short*)(ws + 262144);   // 3*128*128*2 = 98304
  float* b1 = (float*)(ws + 360448);                      // 256*4
  float* b2 = (float*)(ws + 361472);                      // 384*4
  float* s0 = (float*)(ws + 363008);                      // 512*4
  float* t0 = (float*)(ws + 365056);                      // 512*4

  tabnet_prep_w1<<<512, 256, 0, stream>>>(shW, sh_g, sh_b, sh_m, sh_v,
                                          bn0_g, bn0_b, bn0_m, bn0_v,
                                          W1t, b1, s0, t0);
  tabnet_prep_w2<<<192, 256, 0, stream>>>(stW, st_g, st_b, st_m, st_v,
                                          W2t, b2);
  tabnet_main<<<1024, 256, SMEM_BYTES, stream>>>(x, s0, t0, W1t, b1,
                                                 W2t, b2, fW, fb, out);
  (void)in_sizes; (void)n_in; (void)out_size; (void)ws_size;
}
